// STN_RNN_18897856103011
// MI455X (gfx1250) — matrix operand order repaired and checked by reading, not compiled, not run on hardware
//
#include <hip/hip_runtime.h>
#include <hip/hip_bf16.h>

// ---------------------------------------------------------------------------
// STN Euler-LSTM recurrence, fused persistent-kernel design for MI455X
//   B=64, T=4096, IN=64, H=512, K=0.5
//   gates = [x_t | h] @ Wc^T + b,  Wc = [W_ih | W_hh]  (2048 x 576)
//   16 persistent workgroups (one 32-col slice of h each), bf16 WMMA
//   h/c state kept in registers in WMMA C/D fragment layout.
// ---------------------------------------------------------------------------

typedef __attribute__((ext_vector_type(16))) __bf16 v16bf;
typedef __attribute__((ext_vector_type(8)))  float  v8f;

#define B_   64
#define T_   4096
#define IN_  64
#define H_   512
#define KF   0.5f
#define NWG  16
#define COLS_PER_WG 32           // H / NWG
#define KTILES 18                // (IN + H) / 32
#define NTILES 8                 // 128 gate cols per WG / 16

// workspace layout (bytes)
#define O_CTR   0                // 2 x unsigned (barrier counter, flag)
#define O_BIAS  4096             // 2048 f32 combined bias
#define O_HBUF  16384            // 2 * 64*512 bf16 double buffer = 131072 B
#define O_WSW   147456           // 16*18*8*512 bf16 swizzled weights = 2359296 B

union AFrag { v16bf v; uint4 q[2]; __bf16 e[16]; };
union BFrag { v16bf v; uint4 q[2]; };

__device__ __forceinline__ float sigm_f(float x) {
  return 1.0f / (1.0f + __expf(-x));
}
__device__ __forceinline__ float tanh_f(float x) {
  float e = __expf(2.0f * x);
  return 1.0f - 2.0f / (e + 1.0f);
}

// device-wide sense-free barrier (monotonic counter + flag), ws-resident
__device__ __forceinline__ void grid_barrier(unsigned* ctr, unsigned target) {
  __threadfence();
  __syncthreads();
  if (threadIdx.x == 0) {
    unsigned arrived = __hip_atomic_fetch_add(&ctr[0], 1u, __ATOMIC_ACQ_REL,
                                              __HIP_MEMORY_SCOPE_AGENT);
    if (arrived == target - 1u) {
      __hip_atomic_store(&ctr[1], target, __ATOMIC_RELEASE,
                         __HIP_MEMORY_SCOPE_AGENT);
    } else {
      while (__hip_atomic_load(&ctr[1], __ATOMIC_ACQUIRE,
                               __HIP_MEMORY_SCOPE_AGENT) < target) {
        __builtin_amdgcn_s_sleep(2);
      }
    }
  }
  __syncthreads();
}

// --------------------------- prep kernels ----------------------------------

__global__ void stn_prep_init(const float* __restrict__ bih,
                              const float* __restrict__ bhh,
                              float* __restrict__ bias,
                              unsigned* __restrict__ ctr,
                              unsigned* __restrict__ hbuf_u32) {
  int i = blockIdx.x * blockDim.x + threadIdx.x;
  int n = gridDim.x * blockDim.x;
  if (i < 2) ctr[i] = 0u;
  if (i < 4 * H_) bias[i] = bih[i] + bhh[i];
  for (int j = i; j < (2 * B_ * H_) / 2; j += n) hbuf_u32[j] = 0u;  // 32768 u32
}

// Swizzle Wc = [W_ih | W_hh] (fp32) into bf16 B-fragment tiles:
// per WG the 128 gate columns are ordered [i(32) f(32) g(32) o(32)];
// each 32x16 (KxN) tile is stored as 32 lanes x 16 bf16 contiguous.
__global__ void stn_prep_w(const float* __restrict__ Wih,
                           const float* __restrict__ Whh,
                           __bf16* __restrict__ wsw) {
  int idx = blockIdx.x * blockDim.x + threadIdx.x;
  const int total = NWG * KTILES * NTILES * 512;
  if (idx >= total) return;
  int j    = idx & 15;
  int lane = (idx >> 4) & 31;
  int tile = idx >> 9;
  int nt = tile & 7;
  int kt = (tile >> 3) % KTILES;
  int wg = tile / (NTILES * KTILES);
  int n_local = nt * 16 + (lane & 15);                 // 0..127
  int grow = (n_local >> 5) * H_ + wg * COLS_PER_WG + (n_local & 31);
  int kg = kt * 32 + ((j >> 3) << 4) + ((lane >> 4) << 3) + (j & 7);  // 0..575
  float w = (kg < IN_) ? Wih[(size_t)grow * IN_ + kg]
                       : Whh[(size_t)grow * H_ + (kg - IN_)];
  wsw[idx] = (__bf16)w;
}

// --------------------------- recurrent kernel ------------------------------

__global__ __launch_bounds__(128, 1)
void stn_lstm_persistent(const float* __restrict__ x,
                         const float* __restrict__ bias,
                         const __bf16* __restrict__ wsw,
                         __bf16* __restrict__ hbuf,
                         unsigned* __restrict__ ctr,
                         float* __restrict__ out) {
  const int wg    = blockIdx.x;            // 0..15  (h column slice)
  const int wave  = threadIdx.x >> 5;      // 0..3   (M tile: batch rows)
  const int lane  = threadIdx.x & 31;
  const int lhalf = lane >> 4;             // 0/1
  const int l15   = lane & 15;
  const int m     = wave * 16 + l15;       // A-fragment row (batch index)

  // per-lane bias scalars for the 8 accumulator tiles (constant over t)
  float bias_s[8];
#pragma unroll
  for (int nt = 0; nt < 8; ++nt) {
    int grow = (nt >> 1) * H_ + wg * COLS_PER_WG + ((nt & 1) << 4) + l15;
    bias_s[nt] = bias[grow];
  }

  // persistent h/c state in C/D fragment layout: [nt 0..1][vgpr 0..7]
  float hreg[2][8], creg[2][8];
#pragma unroll
  for (int nt = 0; nt < 2; ++nt)
#pragma unroll
    for (int v = 0; v < 8; ++v) { hreg[nt][v] = 0.0f; creg[nt][v] = 0.0f; }

#pragma unroll 1
  for (int t = 0; t < T_; ++t) {
    const __bf16* __restrict__ hprev = hbuf + (size_t)(t & 1) * (B_ * H_);
    __bf16* __restrict__ hnext = hbuf + (size_t)((t & 1) ^ 1) * (B_ * H_);

    v8f acc[8];
#pragma unroll
    for (int nt = 0; nt < 8; ++nt)
#pragma unroll
      for (int v = 0; v < 8; ++v) acc[nt][v] = bias_s[nt];

    // --- K tiles 0..1: x_t (fp32 -> bf16 on the fly) ---
    const float* xr = x + ((size_t)m * T_ + t) * IN_;
#pragma unroll
    for (int kt = 0; kt < 2; ++kt) {
      const int k0 = kt * 32 + lhalf * 8;
      float4 f0 = *(const float4*)(xr + k0);
      float4 f1 = *(const float4*)(xr + k0 + 4);
      float4 f2 = *(const float4*)(xr + k0 + 16);
      float4 f3 = *(const float4*)(xr + k0 + 20);
      float xf[16] = {f0.x, f0.y, f0.z, f0.w, f1.x, f1.y, f1.z, f1.w,
                      f2.x, f2.y, f2.z, f2.w, f3.x, f3.y, f3.z, f3.w};
      AFrag af;
#pragma unroll
      for (int j = 0; j < 16; ++j) af.e[j] = (__bf16)xf[j];

      const __bf16* bt =
          wsw + (size_t)((wg * KTILES + kt) * NTILES) * 512 + lane * 16;
#pragma unroll
      for (int nt = 0; nt < 8; ++nt) {
        BFrag bf;
        const uint4* bp = (const uint4*)(bt + nt * 512);
        bf.q[0] = bp[0]; bf.q[1] = bp[1];
        acc[nt] = __builtin_amdgcn_wmma_f32_16x16x32_bf16(
            false, af.v, false, bf.v, (short)0, acc[nt], false, false);
      }
    }

    // --- K tiles 2..17: previous h (bf16 double buffer, L2-resident) ---
#pragma unroll 2
    for (int kt = 2; kt < KTILES; ++kt) {
      const int kh = kt * 32 - IN_ + lhalf * 8;
      const __bf16* hr = hprev + (size_t)m * H_;
      AFrag af;
      af.q[0] = *(const uint4*)(hr + kh);
      af.q[1] = *(const uint4*)(hr + kh + 16);

      const __bf16* bt =
          wsw + (size_t)((wg * KTILES + kt) * NTILES) * 512 + lane * 16;
      __builtin_prefetch((const void*)(bt + NTILES * 512), 0, 0);
#pragma unroll
      for (int nt = 0; nt < 8; ++nt) {
        BFrag bf;
        const uint4* bp = (const uint4*)(bt + nt * 512);
        bf.q[0] = bp[0]; bf.q[1] = bp[1];
        acc[nt] = __builtin_amdgcn_wmma_f32_16x16x32_bf16(
            false, af.v, false, bf.v, (short)0, acc[nt], false, false);
      }
    }

    // --- activations + Euler state update + stores ---
#pragma unroll
    for (int nt = 0; nt < 2; ++nt) {
      const int hcol = wg * COLS_PER_WG + nt * 16 + l15;
#pragma unroll
      for (int v = 0; v < 8; ++v) {
        const int b = wave * 16 + lhalf * 8 + v;
        float iv = sigm_f(acc[0 + nt][v]);
        float fv = sigm_f(acc[2 + nt][v]);
        float gv = tanh_f(acc[4 + nt][v]);
        float ov = sigm_f(acc[6 + nt][v]);
        float c_old = creg[nt][v], h_old = hreg[nt][v];
        float cn = fv * c_old + iv * gv;
        float hn = ov * tanh_f(cn);
        float h2 = h_old + KF * (hn - h_old);
        float c2 = c_old + KF * (cn - c_old);
        hreg[nt][v] = h2; creg[nt][v] = c2;
        hnext[(size_t)b * H_ + hcol] = (__bf16)h2;
        out[((size_t)b * T_ + t) * H_ + hcol] = h2;
      }
    }

    grid_barrier(ctr, (unsigned)(t + 1) * NWG);
  }

  // final = concat(hT, cT)  -> out[B*T*H ..]
  const size_t SBASE = (size_t)B_ * T_ * H_;
#pragma unroll
  for (int nt = 0; nt < 2; ++nt) {
    const int hcol = wg * COLS_PER_WG + nt * 16 + l15;
#pragma unroll
    for (int v = 0; v < 8; ++v) {
      const int b = wave * 16 + lhalf * 8 + v;
      out[SBASE + (size_t)b * (2 * H_) + hcol]      = hreg[nt][v];
      out[SBASE + (size_t)b * (2 * H_) + H_ + hcol] = creg[nt][v];
    }
  }
}

// --------------------------- launch ----------------------------------------

extern "C" void kernel_launch(void* const* d_in, const int* in_sizes, int n_in,
                              void* d_out, int out_size, void* d_ws,
                              size_t ws_size, hipStream_t stream) {
  (void)in_sizes; (void)n_in; (void)out_size; (void)ws_size;
  const float* x   = (const float*)d_in[0];
  const float* Wih = (const float*)d_in[1];
  const float* Whh = (const float*)d_in[2];
  const float* bih = (const float*)d_in[3];
  const float* bhh = (const float*)d_in[4];
  float* out = (float*)d_out;

  char* ws = (char*)d_ws;
  unsigned* ctr  = (unsigned*)(ws + O_CTR);
  float*    bias = (float*)(ws + O_BIAS);
  __bf16*   hbuf = (__bf16*)(ws + O_HBUF);
  __bf16*   wsw  = (__bf16*)(ws + O_WSW);

  stn_prep_init<<<64, 256, 0, stream>>>(bih, bhh, bias, ctr, (unsigned*)hbuf);

  const int wtot = NWG * KTILES * NTILES * 512;  // 1,179,648
  stn_prep_w<<<(wtot + 255) / 256, 256, 0, stream>>>(Wih, Whh, wsw);

  stn_lstm_persistent<<<NWG, 128, 0, stream>>>(x, bias, wsw, hbuf, ctr, out);
}
